// LambdaLayer_1013612282093
// MI455X (gfx1250) — matrix operand order repaired and checked by reading, not compiled, not run on hardware
//
#include <hip/hip_runtime.h>
#include <math.h>

typedef __bf16 bf16;
typedef __attribute__((ext_vector_type(16))) __bf16 v16bf;
typedef __attribute__((ext_vector_type(8)))  __bf16 v8bf;
typedef __attribute__((ext_vector_type(8)))  float  v8f;
typedef __attribute__((ext_vector_type(4)))  int    v4i;
typedef __attribute__((address_space(1))) v4i* gv4i_p;   // global int4*
typedef __attribute__((address_space(3))) v4i* lv4i_p;   // LDS int4*

#define NB   16
#define NC   256
#define NH   56
#define NW   56
#define NN   3136          // 56*56
#define DU   4
#define DV   64
#define KSC   30           // conv K-steps: u*15 rows x 16 (dx padded 15->16) = 960
#define EPSV 1e-5f

#if defined(__gfx1250__) && __has_builtin(__builtin_amdgcn_global_load_async_to_lds_b128)
#define USE_ASYNC 1
#else
#define USE_ASYNC 0
#endif

// ---- WMMA fragment index helpers (ISA 7.12.2, 16-bit A 16x32 / B 32x16) ----
__device__ __forceinline__ int a_kidx(int e, int g) {
    int j = e >> 1, p = e & 1;
    int base = (j < 4) ? (2 * j) : (16 + 2 * (j - 4));
    return base + 8 * g + p;       // two contiguous 8-runs: [8g,8g+8) and [16+8g,16+8g+8)
}
__device__ __forceinline__ int b_kidx(int e, int g) {
    return 16 * g + e;             // one contiguous 16-run per half-wave
}
__device__ __forceinline__ bf16 f2b(float f) { return (bf16)f; }

__device__ __forceinline__ v8f wmma_bf16(v16bf a, v16bf b, v8f c) {
    return __builtin_amdgcn_wmma_f32_16x16x32_bf16(false, a, false, b, (short)0, c, false, false);
}

// --------------------------------------- transpose+convert x -> xt[b][n][c]
__global__ __launch_bounds__(256) void k_cvt_t(const float* __restrict__ x,
                                               bf16* __restrict__ xt) {
    __shared__ float tile[32][33];
    const int b = blockIdx.z, c0 = blockIdx.y * 32, n0 = blockIdx.x * 32;
    for (int i = threadIdx.x; i < 1024; i += 256) {
        int cl = i >> 5, nl = i & 31;
        tile[cl][nl] = x[((size_t)b * NC + c0 + cl) * NN + n0 + nl];
    }
    __syncthreads();
    for (int i = threadIdx.x; i < 1024; i += 256) {
        int nl = i >> 5, cl = i & 31;
        xt[((size_t)b * NN + n0 + nl) * NC + c0 + cl] = f2b(tile[cl][nl]);
    }
}

// ---------------- pre-pack q/k/v weights into WMMA A-fragment order --------
// Apack[((dt*8+ks)*32+lane)*16+e] = W[(ks*32+b_kidx(e,g)) * ncols + col0 + ln]
__global__ __launch_bounds__(256) void k_wpack(const float* __restrict__ Wq,
                                               const float* __restrict__ Wk,
                                               const float* __restrict__ Wv,
                                               bf16* __restrict__ Apack) {
    int i = blockIdx.x * 256 + threadIdx.x;          // exactly 24*8*32*16
    int e = i & 15, lane = (i >> 4) & 31, ks = (i >> 9) & 7, dt = i >> 12;
    int g = lane >> 4, ln = lane & 15;
    const float* W; int ncols, col0;
    if (dt < 4)      { W = Wq; ncols = 64;  col0 = dt * 16; }
    else if (dt < 8) { W = Wk; ncols = 64;  col0 = (dt - 4) * 16; }
    else             { W = Wv; ncols = 256; col0 = (dt - 8) * 16; }
    int c = ks * 32 + b_kidx(e, g);
    Apack[i] = f2b(W[(size_t)c * ncols + col0 + ln]);
}

// ------- pre-pack conv weights, dx padded 15->16, A-fragment order ---------
// K index k = row*16 + dx, row = u*15+dy (0..59), dx 0..15 (15 => zero)
__global__ __launch_bounds__(256) void k_cpack(const float* __restrict__ pos_w,
                                               bf16* __restrict__ Ac) {
    const int stride = gridDim.x * 256;
    for (int i = blockIdx.x * 256 + threadIdx.x; i < KSC * 32 * 16; i += stride) {
        int e = i & 15, lane = (i >> 4) & 31, ks = i >> 9;
        int g = lane >> 4, m = lane & 15;
        int k = ks * 32 + b_kidx(e, g);
        int row = k >> 4, dx = k & 15;
        Ac[i] = (dx < 15) ? f2b(pos_w[m * 900 + row * 15 + dx]) : f2b(0.f);
    }
}

// ------------------------------------------------------- fused q/k/v proj
__global__ __launch_bounds__(128) void k_proj(
    const bf16* __restrict__ xt, const bf16* __restrict__ Apack,
    const float* __restrict__ gq, const float* __restrict__ bq,
    const float* __restrict__ mq, const float* __restrict__ vq,
    const float* __restrict__ gv, const float* __restrict__ bv,
    const float* __restrict__ mv, const float* __restrict__ vv,
    bf16* __restrict__ qb, float* __restrict__ kf, bf16* __restrict__ vb)
{
    __shared__ __align__(32) bf16 sA[8 * 32 * 16];   // this dt's packed A (8KB)
    const int b  = blockIdx.z;
    const int dt = blockIdx.y;                       // 0..23
    const int nt = blockIdx.x;                       // 0..48
    const int wave = threadIdx.x >> 5;
    const int lane = threadIdx.x & 31;
    const int g = lane >> 4, ln = lane & 15;
    const int n0 = nt * 64 + wave * 16;

    for (int i = threadIdx.x; i < 512; i += 128)
        ((uint4*)sA)[i] = ((const uint4*)(Apack + (size_t)dt * 4096))[i];
    __syncthreads();

    const bf16* xrow = xt + ((size_t)b * NN + n0 + ln) * NC + 16 * g;

    v8f acc = {};
    #pragma unroll
    for (int ks = 0; ks < 8; ++ks) {
        v16bf af  = *(const v16bf*)&sA[(ks * 32 + lane) * 16];
        v16bf bfv = *(const v16bf*)(xrow + ks * 32);
        acc = wmma_bf16(af, bfv, acc);
    }

    #pragma unroll
    for (int r = 0; r < 8; ++r) {
        const int m = r + 8 * g;
        const int n = n0 + ln;
        const int d = dt * 16 + m;                   // 0..383
        float val = acc[r];
        if (d < 64) {                                // q + BN
            float inv = gq[d] * rsqrtf(vq[d] + EPSV);
            val = val * inv + (bq[d] - mq[d] * inv);
            qb[((size_t)b * 64 + d) * NN + n] = f2b(val);
        } else if (d < 128) {                        // raw k (f32 for softmax)
            kf[((size_t)b * 64 + (d - 64)) * NN + n] = val;
        } else {                                     // v + BN
            int ch = d - 128;
            float inv = gv[ch] * rsqrtf(vv[ch] + EPSV);
            val = val * inv + (bv[ch] - mv[ch] * inv);
            vb[((size_t)b * 256 + ch) * NN + n] = f2b(val);
        }
    }
}

// --------------------------------------------------------------- softmax(k)
__global__ __launch_bounds__(256) void k_softmax(const float* __restrict__ kf,
                                                 bf16* __restrict__ ksm)
{
    const int row = blockIdx.x;                      // b*64 + (u*16+k)
    const float* p = kf + (size_t)row * NN;
    const float4* p4 = (const float4*)p;
    __shared__ float red[256];
    const int t = threadIdx.x;

    float mx = -INFINITY;
    for (int i = t; i < NN / 4; i += 256) {
        float4 v = p4[i];
        mx = fmaxf(mx, fmaxf(fmaxf(v.x, v.y), fmaxf(v.z, v.w)));
    }
    red[t] = mx; __syncthreads();
    for (int s = 128; s > 0; s >>= 1) { if (t < s) red[t] = fmaxf(red[t], red[t + s]); __syncthreads(); }
    mx = red[0]; __syncthreads();

    float sum = 0.f;
    for (int i = t; i < NN / 4; i += 256) {
        float4 v = p4[i];
        sum += __expf(v.x - mx) + __expf(v.y - mx) + __expf(v.z - mx) + __expf(v.w - mx);
    }
    red[t] = sum; __syncthreads();
    for (int s = 128; s > 0; s >>= 1) { if (t < s) red[t] += red[t + s]; __syncthreads(); }
    const float inv = 1.0f / red[0];

    bf16* o = ksm + (size_t)row * NN;
    for (int i = t; i < NN; i += 256) o[i] = f2b(__expf(p[i] - mx) * inv);
}

// ------------------------------------------- Lc[b,k,v] = sum_u,m k*v (WMMA)
__global__ __launch_bounds__(128) void k_lc(const bf16* __restrict__ ksm,
                                            const bf16* __restrict__ vb,
                                            float* __restrict__ Lc)
{
    const int b    = blockIdx.x;
    const int wave = threadIdx.x >> 5;
    const int lane = threadIdx.x & 31;
    const int g = lane >> 4, ln = lane & 15;
    const int v0 = wave * 16;

    v8f acc = {};
    for (int u = 0; u < DU; ++u) {
        const bf16* kp = ksm + ((size_t)(b * 4 + u) * 16 + ln) * NN;          // row k=ln
        const bf16* vp = vb  + ((size_t)b * 256 + u * 64 + v0 + ln) * NN;     // row v
        for (int ks = 0; ks < NN / 32; ++ks) {
            const int m0 = ks * 32;
            v8bf lo = *(const v8bf*)(kp + m0 + 16 * g);
            v8bf hi = *(const v8bf*)(kp + m0 + 16 * g + 8);
            v16bf af = __builtin_shufflevector(lo, hi, 0,1,2,3,4,5,6,7,
                                               8,9,10,11,12,13,14,15);
            v16bf bfv = *(const v16bf*)(vp + m0 + 16 * g);
            acc = wmma_bf16(af, bfv, acc);
        }
    }
    #pragma unroll
    for (int r = 0; r < 8; ++r)
        Lc[((size_t)b * 16 + r + 8 * g) * 64 + v0 + ln] = acc[r];
}

// -------------------------- position lambda: implicit-GEMM 15x15 conv (WMMA)
// K = 960 (60 rows x 16 padded dx). Row for half-wave g at step ks: ks*2+g.
// B fragment = 16 contiguous LDS entries starting at row*32 + ln.
__global__ __launch_bounds__(128) void k_conv(const float* __restrict__ pos_b,
                                              const bf16* __restrict__ Ac,
                                              const bf16* __restrict__ vb,
                                              bf16* __restrict__ Lp)
{
    __shared__ __align__(32) bf16 sAc[KSC * 32 * 16];   // 30720 B
    __shared__ __align__(32) bf16 s_patch[4][1952];     // per-wave [60 rows][32]
    __shared__ float s_pb[16];

    const int b  = blockIdx.z;
    const int y  = blockIdx.y;
    const int x0 = blockIdx.x * 16;
    const int wave = threadIdx.x >> 5;
    const int lane = threadIdx.x & 31;
    const int g = lane >> 4, ln = lane & 15;

    for (int i = threadIdx.x; i < KSC * 32 * 16 * 2 / 16; i += 128)
        ((uint4*)sAc)[i] = ((const uint4*)Ac)[i];
    if (threadIdx.x < 16) s_pb[threadIdx.x] = pos_b[threadIdx.x];
    __syncthreads();

    const bf16* vbb = vb + (size_t)b * 256 * NN;

    for (int v16 = 0; v16 < 16; ++v16) {
        const int v = wave * 16 + v16;
        // stage zero-padded patch: row = u*15+dy (0..59), 30 valid cols
        for (int i = lane; i < 1920; i += 32) {
            int row = i >> 5, xx = i & 31;
            int u = row / 15, dyr = row % 15;
            int yy = y + dyr - 7, gx = x0 + xx - 7;
            bf16 val = f2b(0.f);
            if (xx < 30 && yy >= 0 && yy < NH && gx >= 0 && gx < NW)
                val = vbb[(size_t)(u * 64 + v) * NN + yy * NW + gx];
            s_patch[wave][i] = val;
        }
        if (lane == 0 && v16 < 15)                       // prefetch next v plane
            __builtin_prefetch(vbb + (size_t)(wave * 16 + v16 + 1) * NN + y * NW, 0, 0);
        __syncthreads();

        v8f acc = {};
        for (int ks = 0; ks < KSC; ++ks) {
            v16bf af = *(const v16bf*)&sAc[(ks * 32 + lane) * 16];
            v16bf bfv;
            __builtin_memcpy(&bfv, &s_patch[wave][(ks * 2 + g) * 32 + ln], 32);
            acc = wmma_bf16(af, bfv, acc);
        }
        __syncthreads();

        const int xc = x0 + ln;
        if (xc < NW) {
            #pragma unroll
            for (int r = 0; r < 8; ++r) {
                const int krow = r + 8 * g;
                Lp[(((size_t)b * 16 + krow) * 64 + v) * NN + y * NW + xc]
                    = f2b(acc[r] + s_pb[krow]);
            }
        }
    }
}

// ------------------------------- combine: out = q . (Lc + Lp), with layout
__global__ __launch_bounds__(256) void k_combine(const bf16* __restrict__ qb,
                                                 const float* __restrict__ Lc,
                                                 const bf16* __restrict__ Lp,
                                                 float* __restrict__ out)
{
    __shared__ __align__(16) bf16 sLp[16 * 64 * 16];    // [k*64+v][n_local]
    __shared__ float sLc[1024];
    __shared__ bf16  sQ[64 * 16];                       // [d][n_local]
    const int b  = blockIdx.y;
    const int n0 = blockIdx.x * 16;

#if USE_ASYNC
    for (int i = threadIdx.x; i < 2048; i += 256) {     // 2048 x 16B async copies
        int kv = i >> 1, half = i & 1;
        const bf16* src = Lp + ((size_t)b * 1024 + kv) * NN + n0 + half * 8;
        bf16* dst = &sLp[kv * 16 + half * 8];
        __builtin_amdgcn_global_load_async_to_lds_b128(
            (gv4i_p)(const void*)src, (lv4i_p)(void*)dst, 0, 0);
    }
#else
    for (int i = threadIdx.x; i < 16 * 64 * 16; i += 256) {
        int nl = i & 15, kv = i >> 4;
        sLp[i] = Lp[((size_t)b * 1024 + kv) * NN + n0 + nl];
    }
#endif
    for (int i = threadIdx.x; i < 1024; i += 256) sLc[i] = Lc[(size_t)b * 1024 + i];
    for (int i = threadIdx.x; i < 1024; i += 256) {
        int d = i >> 4, nl = i & 15;
        sQ[i] = qb[((size_t)b * 64 + d) * NN + n0 + nl];
    }
#if USE_ASYNC
#if __has_builtin(__builtin_amdgcn_s_wait_asynccnt)
    __builtin_amdgcn_s_wait_asynccnt(0);
#else
    asm volatile("s_wait_asynccnt 0" ::: "memory");
#endif
#endif
    __syncthreads();

    const int n_l = threadIdx.x & 15;
    const int og  = threadIdx.x >> 4;
    const int o0  = og * 16;
    const int h   = o0 >> 6;
    const int v0  = o0 & 63;

    float qv[16];
    #pragma unroll
    for (int k = 0; k < 16; ++k) qv[k] = (float)sQ[(h * 16 + k) * 16 + n_l];

    float acc[16] = {};
    #pragma unroll 4
    for (int k = 0; k < 16; ++k) {
        const float qk = qv[k];
        #pragma unroll
        for (int j = 0; j < 16; ++j) {
            const int vv_ = v0 + j;
            acc[j] += qk * (sLc[k * 64 + vv_] + (float)sLp[(k * 64 + vv_) * 16 + n_l]);
        }
    }
    #pragma unroll
    for (int j = 0; j < 16; ++j)
        out[((size_t)b * 256 + o0 + j) * NN + n0 + n_l] = acc[j];
}

// --------------------------------------------------------------------- host
extern "C" void kernel_launch(void* const* d_in, const int* in_sizes, int n_in,
                              void* d_out, int out_size, void* d_ws, size_t ws_size,
                              hipStream_t stream) {
    const float* x     = (const float*)d_in[0];
    const float* Wq    = (const float*)d_in[1];
    const float* Wk    = (const float*)d_in[2];
    const float* Wv    = (const float*)d_in[3];
    const float* pos_w = (const float*)d_in[4];
    const float* pos_b = (const float*)d_in[5];
    const float* gq    = (const float*)d_in[6];
    const float* bq    = (const float*)d_in[7];
    const float* mq    = (const float*)d_in[8];
    const float* vq    = (const float*)d_in[9];
    const float* gv    = (const float*)d_in[10];
    const float* bv    = (const float*)d_in[11];
    const float* mv    = (const float*)d_in[12];
    const float* vv    = (const float*)d_in[13];
    float* out = (float*)d_out;

    char* ws = (char*)d_ws;
    size_t off = 0;
    auto alloc = [&](size_t bytes) { size_t o = off; off = (off + bytes + 255) & ~(size_t)255; return o; };
    bf16*  xt    = (bf16*) (ws + alloc((size_t)NB * NC * NN * 2));
    bf16*  qb    = (bf16*) (ws + alloc((size_t)NB * 64 * NN * 2));
    float* kf    = (float*)(ws + alloc((size_t)NB * 64 * NN * 4));
    bf16*  ksm   = (bf16*) (ws + alloc((size_t)NB * 64 * NN * 2));
    bf16*  vb    = (bf16*) (ws + alloc((size_t)NB * 256 * NN * 2));
    float* Lc    = (float*)(ws + alloc((size_t)NB * 16 * 64 * 4));
    bf16*  Lp    = (bf16*) (ws + alloc((size_t)NB * 16 * 64 * NN * 2));
    bf16*  Apack = (bf16*) (ws + alloc((size_t)24 * 8 * 32 * 16 * 2));
    bf16*  Ac    = (bf16*) (ws + alloc((size_t)KSC * 32 * 16 * 2));

    k_cvt_t<<<dim3(NN / 32, NC / 32, NB), 256, 0, stream>>>(x, xt);
    k_wpack<<<384, 256, 0, stream>>>(Wq, Wk, Wv, Apack);
    k_cpack<<<64, 256, 0, stream>>>(pos_w, Ac);

    k_proj<<<dim3(49, 24, NB), 128, 0, stream>>>(xt, Apack,
                                                 gq, bq, mq, vq, gv, bv, mv, vv,
                                                 qb, kf, vb);

    k_softmax<<<NB * 64, 256, 0, stream>>>(kf, ksm);
    k_lc<<<NB, 128, 0, stream>>>(ksm, vb, Lc);
    k_conv<<<dim3(4, NH, NB), 128, 0, stream>>>(pos_b, Ac, vb, Lp);
    k_combine<<<dim3(NN / 16, NB), 256, 0, stream>>>(qb, Lc, Lp, out);
}
